// InfiniViT_4277787427208
// MI455X (gfx1250) — matrix operand order repaired and checked by reading, not compiled, not run on hardware
//
#include <hip/hip_runtime.h>
#include <hip/hip_bf16.h>
#include <math.h>

// ---------------------------------------------------------------------------
// InfiniViT block for MI455X (gfx1250, wave32, WMMA bf16 -> f32 accumulate)
//   C=1024, H=16, D=64, T=2048, B=2, M=64, HID=4096
// All GEMMs run on v_wmma_f32_16x16x32_bf16 with double-buffered 64-deep LDS
// tiles filled by GLOBAL_LOAD_ASYNC_TO_LDS_B128 (ASYNCcnt), 8 WMMAs per sync.
// Attention is flash-style (online softmax), never materializing the
// 2048x2048 score matrix. Workspace arena use: ~80.5 MB.
// ---------------------------------------------------------------------------

typedef __attribute__((ext_vector_type(16))) __bf16 v16bf;
typedef __attribute__((ext_vector_type(8)))  float  v8f;

#define WMMA_BF16(a, b, c) \
  __builtin_amdgcn_wmma_f32_16x16x32_bf16(false, (a), false, (b), (short)0, (c), false, false)

__device__ __forceinline__ __bf16 f2bf(float f) {
  union { float f; unsigned u; } in; in.f = f;
  unsigned u = in.u + 0x7FFFu + ((in.u >> 16) & 1u);   // round-to-nearest-even
  union { unsigned short s; __bf16 b; } out; out.s = (unsigned short)(u >> 16);
  return out.b;
}

union FragU { v16bf v; uint4 q[2]; };

// A operand, 16x32 bf16 (M x K), row-major source, ld in elements.
// ISA layout: lanes 0-15 row=lane K{0..7,16..23}; lanes 16-31 row=lane-16 K{8..15,24..31}.
__device__ __forceinline__ v16bf load_afrag(const __bf16* base, int ld) {
  int lane = threadIdx.x & 31;
  const __bf16* p = base + (size_t)(lane & 15) * ld + ((lane >> 4) << 3);
  FragU f;
  f.q[0] = *(const uint4*)p;
  f.q[1] = *(const uint4*)(p + 16);
  return f.v;
}

// B operand, 32x16 bf16 (K x N), sourced from N-major ("Bt[n][k]") memory.
// ISA layout: lanes 0-15 col=lane K{0..7,16..23}; lanes 16-31 col=lane-16 K{8..15,24..31}.
__device__ __forceinline__ v16bf load_bfrag(const __bf16* base, int ld) {
  int lane = threadIdx.x & 31;
  const __bf16* p = base + (size_t)(lane & 15) * ld + ((lane >> 4) << 3);
  FragU f;
  f.q[0] = *(const uint4*)p;
  f.q[1] = *(const uint4*)(p + 16);
  return f.v;
}

// A-operand fragment from per-wave LDS P-tile (16 x 32, row stride 32).
__device__ __forceinline__ v16bf load_pfrag(const __bf16 (&psc)[16][32]) {
  int lane = threadIdx.x & 31;
  const __bf16* p = &psc[lane & 15][(lane >> 4) << 3];
  FragU f;
  f.q[0] = *(const uint4*)p;
  f.q[1] = *(const uint4*)(p + 16);
  return f.v;
}

// ---- CDNA5 async global->LDS copy (ASYNCcnt path) --------------------------
// GV mode: the LDS destination address comes from the VDST VGPR; generic shared pointers carry the
// LDS byte offset in their low 32 bits.
__device__ __forceinline__ void async_copy16(void* lds_dst, const void* gsrc) {
  asm volatile("global_load_async_to_lds_b128 %0, %1, off"
               :: "v"((unsigned)(uintptr_t)lds_dst),
                  "v"((unsigned long long)(uintptr_t)gsrc)
               : "memory");
}
__device__ __forceinline__ void async_wait0() {
  asm volatile("s_wait_asynccnt 0x0" ::: "memory");
}

// ---------------------------------------------------------------------------
// Generic WMMA GEMM: D[m,n] = epi(sum_k A[m,k] * Bt[n,k])
// A: Mrows x K row-major bf16 ; Bt: Ncols x K row-major bf16.
// Block: 256 threads = 8 waves as 2x4 -> 64 x 128 output tile; wave = 32x32.
// K-steps of 64 are staged in LDS (double-buffered, 48 KB) by async LDS loads
// that overlap the 8 WMMAs of the previous step. K is compile-time so the
// k-loop unrolls/pipelines.
// grid.x = N/128, grid.y = M/64. K multiple of 64, M mult of 64, N mult of 128.
// ---------------------------------------------------------------------------
template <int K, typename Epi>
__global__ __launch_bounds__(256) void gemm_bf16(const __bf16* __restrict__ A,
                                                 const __bf16* __restrict__ Bt,
                                                 Epi epi) {
  __shared__ __bf16 sA[2][64][64];    //  8 KB per buffer
  __shared__ __bf16 sB[2][128][64];   // 16 KB per buffer
  int tid = threadIdx.x;
  int wave = tid >> 5;
  int lane = tid & 31;
  int rowBlk = blockIdx.y * 64;
  int colBlk = blockIdx.x * 128;
  int row0 = (wave >> 2) * 32;
  int col0 = (wave & 3) * 32;

  // 16-byte chunk coordinates: 256 threads move A (64x64) with 2 chunks each
  // and B (128x64) with 4 chunks each.
  int cr = tid >> 2;            // 0..63
  int co = (tid & 3) << 3;      // element offset 0,8,16,24

  auto stage = [&](int buf, int k) {
    const __bf16* ga = A + (size_t)(rowBlk + cr) * K + k + co;
    async_copy16(&sA[buf][cr][co],      ga);
    async_copy16(&sA[buf][cr][co + 32], ga + 32);
    const __bf16* gb0 = Bt + (size_t)(colBlk + cr) * K + k + co;
    const __bf16* gb1 = Bt + (size_t)(colBlk + cr + 64) * K + k + co;
    async_copy16(&sB[buf][cr][co],           gb0);
    async_copy16(&sB[buf][cr][co + 32],      gb0 + 32);
    async_copy16(&sB[buf][cr + 64][co],      gb1);
    async_copy16(&sB[buf][cr + 64][co + 32], gb1 + 32);
  };

  v8f acc00 = {}, acc01 = {}, acc10 = {}, acc11 = {};
  const int nk = K >> 6;
  stage(0, 0);
#pragma unroll 2
  for (int i = 0; i < nk; ++i) {
    async_wait0();        // this wave's copies for stage i are in LDS
    __syncthreads();      // all waves' copies done; also WAR-protects buf^1
    if (i + 1 < nk) stage((i + 1) & 1, (i + 1) << 6);
    int buf = i & 1;
#pragma unroll
    for (int kc = 0; kc < 64; kc += 32) {
      v16bf fa0 = load_afrag(&sA[buf][row0][kc], 64);
      v16bf fa1 = load_afrag(&sA[buf][row0 + 16][kc], 64);
      v16bf fb0 = load_bfrag(&sB[buf][col0][kc], 64);
      v16bf fb1 = load_bfrag(&sB[buf][col0 + 16][kc], 64);
      acc00 = WMMA_BF16(fa0, fb0, acc00);
      acc01 = WMMA_BF16(fa0, fb1, acc01);
      acc10 = WMMA_BF16(fa1, fb0, acc10);
      acc11 = WMMA_BF16(fa1, fb1, acc11);
    }
  }

  // C layout: VGPR r -> M = r + (lane>=16 ? 8 : 0), N = lane & 15.
  int mlo = (lane >> 4) << 3;
  int n = lane & 15;
#pragma unroll
  for (int r = 0; r < 8; ++r) {
    epi(rowBlk + row0 + r + mlo,      colBlk + col0 + n,      acc00[r]);
    epi(rowBlk + row0 + r + mlo,      colBlk + col0 + 16 + n, acc01[r]);
    epi(rowBlk + row0 + 16 + r + mlo, colBlk + col0 + n,      acc10[r]);
    epi(rowBlk + row0 + 16 + r + mlo, colBlk + col0 + 16 + n, acc11[r]);
  }
}

// ---- Epilogues -------------------------------------------------------------

struct EpiQKV {   // split 3C columns into Q, K (row-major B,H,T,D) and V^T (B,H,D,T)
  const float* bias;
  __bf16 *q, *k, *vt;
  __device__ void operator()(int m, int n, float v) const {
    v += bias[n];
    int which = n >> 10;         // 0:q 1:k 2:v
    int c = n & 1023;
    int h = c >> 6, d = c & 63;
    int b = m >> 11, t = m & 2047;   // T = 2048
    size_t bh = (size_t)(b * 16 + h);
    __bf16 bv = f2bf(v);
    if (which == 0)      q[(bh * 2048 + t) * 64 + d] = bv;
    else if (which == 1) k[(bh * 2048 + t) * 64 + d] = bv;
    else                 vt[(bh * 64 + d) * 2048 + t] = bv;
  }
};

struct EpiBiasResAdd {  // out[m,1024] = acc + bias[n] + resid[m,1024] (fp32)
  const float* bias;
  const float* resid;
  float* out;
  __device__ void operator()(int m, int n, float v) const {
    size_t i = (size_t)m * 1024 + n;
    out[i] = v + bias[n] + resid[i];
  }
};

struct EpiGelu {  // hid[m,4096] = gelu(acc + bias[n]) as bf16 (exact erf gelu)
  const float* bias;
  __bf16* out;
  __device__ void operator()(int m, int n, float v) const {
    v += bias[n];
    float g = 0.5f * v * (1.0f + erff(v * 0.70710678118654752f));
    out[(size_t)m * 4096 + n] = f2bf(g);
  }
};

// ---------------------------------------------------------------------------
// LayerNorm over C=1024 (one row per block), writes bf16.
// ---------------------------------------------------------------------------
__global__ __launch_bounds__(256) void ln_kernel(const float* __restrict__ x,
                                                 const float* __restrict__ g,
                                                 const float* __restrict__ bta,
                                                 __bf16* __restrict__ out) {
  int row = blockIdx.x;
  const float* xr = x + (size_t)row * 1024;
  float vals[4], s = 0.f, ss = 0.f;
#pragma unroll
  for (int i = 0; i < 4; ++i) {
    float v = xr[threadIdx.x + i * 256];
    vals[i] = v; s += v; ss += v * v;
  }
  for (int m = 16; m >= 1; m >>= 1) { s += __shfl_xor(s, m, 32); ss += __shfl_xor(ss, m, 32); }
  __shared__ float red[2][8];
  if ((threadIdx.x & 31) == 0) { red[0][threadIdx.x >> 5] = s; red[1][threadIdx.x >> 5] = ss; }
  __syncthreads();
  s = 0.f; ss = 0.f;
#pragma unroll
  for (int w = 0; w < 8; ++w) { s += red[0][w]; ss += red[1][w]; }
  float mu = s * (1.f / 1024.f);
  float var = ss * (1.f / 1024.f) - mu * mu;
  float rstd = rsqrtf(var + 1e-5f);
#pragma unroll
  for (int i = 0; i < 4; ++i) {
    int c = threadIdx.x + i * 256;
    out[(size_t)row * 1024 + c] = f2bf((vals[i] - mu) * rstd * g[c] + bta[c]);
  }
}

// w: K x N fp32 row-major -> wt: N x K bf16 row-major.  grid(N/16, K/16), 256 thr.
__global__ __launch_bounds__(256) void transpose_bf16(const float* __restrict__ w,
                                                      __bf16* __restrict__ wt,
                                                      int K, int N) {
  int n = blockIdx.x * 16 + (threadIdx.x & 15);
  int k = blockIdx.y * 16 + (threadIdx.x >> 4);
  wt[(size_t)n * K + k] = f2bf(w[(size_t)k * N + n]);
}

// mem = memory(64x1024) @ w_mem + b_mem; write mem_k[h][m][d] and mem_v^T[h][d][m].
__global__ __launch_bounds__(256) void memproj_kernel(const float* __restrict__ memory,
                                                      const float* __restrict__ w,
                                                      const float* __restrict__ bias,
                                                      __bf16* __restrict__ mem_k,
                                                      __bf16* __restrict__ mem_vt) {
  int m = blockIdx.x;
#pragma unroll
  for (int i = 0; i < 4; ++i) {
    int n = threadIdx.x + i * 256;
    float acc = bias[n];
    for (int k0 = 0; k0 < 1024; ++k0)
      acc += memory[m * 1024 + k0] * w[(size_t)k0 * 1024 + n];
    int h = n >> 6, d = n & 63;
    __bf16 bv = f2bf(acc);
    mem_k[((size_t)h * 64 + m) * 64 + d] = bv;
    mem_vt[((size_t)h * 64 + d) * 64 + m] = bv;
  }
}

// ---------------------------------------------------------------------------
// Flash-attention step: given two 16x16 score tiles (keys kb0..+15, +16..+31)
// in C-layout, do online softmax update and O += P @ V via WMMA.
// P is transposed C-layout -> A-layout through per-wave LDS.
// ---------------------------------------------------------------------------
__device__ __forceinline__ void flash_step(const v8f& s0r, const v8f& s1r, float scale,
                                           float* mst, float* lst, v8f* o,
                                           const __bf16* __restrict__ vt, int ldv, int kb0,
                                           __bf16 (&psc)[16][32]) {
  int lane = threadIdx.x & 31;
  int hi8 = (lane >> 4) << 3;
  int col = lane & 15;
  float a0[8], a1[8], mnew[8];
#pragma unroll
  for (int r = 0; r < 8; ++r) { a0[r] = s0r[r] * scale; a1[r] = s1r[r] * scale; }
#pragma unroll
  for (int r = 0; r < 8; ++r) {   // row max across the 16 lanes holding this row
    float v = fmaxf(a0[r], a1[r]);
    v = fmaxf(v, __shfl_xor(v, 1, 16));
    v = fmaxf(v, __shfl_xor(v, 2, 16));
    v = fmaxf(v, __shfl_xor(v, 4, 16));
    v = fmaxf(v, __shfl_xor(v, 8, 16));
    mnew[r] = fmaxf(mst[r], v);
  }
#pragma unroll
  for (int r = 0; r < 8; ++r) {
    float p0 = __expf(a0[r] - mnew[r]);
    float p1 = __expf(a1[r] - mnew[r]);
    psc[r + hi8][col]      = f2bf(p0);
    psc[r + hi8][col + 16] = f2bf(p1);
    float rs = p0 + p1;
    rs += __shfl_xor(rs, 1, 16);
    rs += __shfl_xor(rs, 2, 16);
    rs += __shfl_xor(rs, 4, 16);
    rs += __shfl_xor(rs, 8, 16);
    float alpha = __expf(mst[r] - mnew[r]);
    lst[r] = lst[r] * alpha + rs;
    mst[r] = mnew[r];
    o[0][r] *= alpha; o[1][r] *= alpha; o[2][r] *= alpha; o[3][r] *= alpha;
  }
  v16bf pf = load_pfrag(psc);   // per-wave scratch: intra-wave DS ordering suffices
#pragma unroll
  for (int j = 0; j < 4; ++j)
    o[j] = WMMA_BF16(pf, load_bfrag(vt + (size_t)(j * 16) * ldv + kb0, ldv), o[j]);
}

// ---------------------------------------------------------------------------
// Attention: local flash attention over T keys + compressive memory attention
// over M=64 slots (independent softmax), summed.  Wave = 16 query rows.
// grid.x = T/128, grid.y = B*H.  Writes attn output bf16 in (B,T,C) layout.
// ---------------------------------------------------------------------------
__global__ __launch_bounds__(256) void attn_kernel(const __bf16* __restrict__ qb,
                                                   const __bf16* __restrict__ kbuf,
                                                   const __bf16* __restrict__ vtb,
                                                   const __bf16* __restrict__ memk,
                                                   const __bf16* __restrict__ memvt,
                                                   __bf16* __restrict__ outb) {
  const int T = 2048, D = 64, C = 1024, MM = 64;
  int wave = threadIdx.x >> 5;
  int lane = threadIdx.x & 31;
  int bh = blockIdx.y;
  int b = bh >> 4, h = bh & 15;
  int t0 = blockIdx.x * 128 + wave * 16;
  __shared__ __bf16 psc[8][16][32];

  const __bf16* q  = qb  + ((size_t)bh * T + t0) * D;
  const __bf16* kk = kbuf + (size_t)bh * T * D;
  const __bf16* vt = vtb + (size_t)bh * D * T;
  v16bf qf0 = load_afrag(q, D);
  v16bf qf1 = load_afrag(q + 32, D);
  const float scale = 0.125f;   // 1/sqrt(64)

  float mst[8], lst[8];
  v8f o[4] = {};
#pragma unroll
  for (int r = 0; r < 8; ++r) { mst[r] = -1e30f; lst[r] = 0.f; }

  for (int kb0 = 0; kb0 < T; kb0 += 32) {           // local attention
    v8f s0 = {}, s1 = {};
    s0 = WMMA_BF16(qf0, load_bfrag(kk + (size_t)kb0 * D, D), s0);
    s0 = WMMA_BF16(qf1, load_bfrag(kk + (size_t)kb0 * D + 32, D), s0);
    s1 = WMMA_BF16(qf0, load_bfrag(kk + (size_t)(kb0 + 16) * D, D), s1);
    s1 = WMMA_BF16(qf1, load_bfrag(kk + (size_t)(kb0 + 16) * D + 32, D), s1);
    flash_step(s0, s1, scale, mst, lst, o, vt, T, kb0, psc[wave]);
  }

  float mst2[8], lst2[8];                            // memory attention
  v8f o2[4] = {};
#pragma unroll
  for (int r = 0; r < 8; ++r) { mst2[r] = -1e30f; lst2[r] = 0.f; }
  const __bf16* mk = memk + (size_t)h * MM * D;
  const __bf16* mv = memvt + (size_t)h * D * MM;
  for (int kb0 = 0; kb0 < MM; kb0 += 32) {
    v8f s0 = {}, s1 = {};
    s0 = WMMA_BF16(qf0, load_bfrag(mk + (size_t)kb0 * D, D), s0);
    s0 = WMMA_BF16(qf1, load_bfrag(mk + (size_t)kb0 * D + 32, D), s0);
    s1 = WMMA_BF16(qf0, load_bfrag(mk + (size_t)(kb0 + 16) * D, D), s1);
    s1 = WMMA_BF16(qf1, load_bfrag(mk + (size_t)(kb0 + 16) * D + 32, D), s1);
    flash_step(s0, s1, scale, mst2, lst2, o2, mv, MM, kb0, psc[wave]);
  }

  int hi8 = (lane >> 4) << 3, col = lane & 15;
#pragma unroll
  for (int r = 0; r < 8; ++r) {
    float i1 = 1.f / lst[r], i2 = 1.f / lst2[r];
    size_t base = ((size_t)(b * T + (t0 + r + hi8))) * C + (size_t)h * D;
    outb[base + col]      = f2bf(o[0][r] * i1 + o2[0][r] * i2);
    outb[base + 16 + col] = f2bf(o[1][r] * i1 + o2[1][r] * i2);
    outb[base + 32 + col] = f2bf(o[2][r] * i1 + o2[2][r] * i2);
    outb[base + 48 + col] = f2bf(o[3][r] * i1 + o2[3][r] * i2);
  }
}

// ---------------------------------------------------------------------------
extern "C" void kernel_launch(void* const* d_in, const int* in_sizes, int n_in,
                              void* d_out, int out_size, void* d_ws, size_t ws_size,
                              hipStream_t stream) {
  (void)in_sizes; (void)n_in; (void)out_size; (void)ws_size;
  const float* x      = (const float*)d_in[0];
  const float* memory = (const float*)d_in[1];
  const float* ln1_g  = (const float*)d_in[2];
  const float* ln1_b  = (const float*)d_in[3];
  const float* w_qkv  = (const float*)d_in[4];
  const float* b_qkv  = (const float*)d_in[5];
  const float* w_out  = (const float*)d_in[6];
  const float* b_out  = (const float*)d_in[7];
  const float* w_mem  = (const float*)d_in[8];
  const float* b_mem  = (const float*)d_in[9];
  const float* ln2_g  = (const float*)d_in[10];
  const float* ln2_b  = (const float*)d_in[11];
  const float* w_fc1  = (const float*)d_in[12];
  const float* b_fc1  = (const float*)d_in[13];
  const float* w_fc2  = (const float*)d_in[14];
  const float* b_fc2  = (const float*)d_in[15];
  float* out = (float*)d_out;

  char* ws = (char*)d_ws;
  const size_t MB = 1024 * 1024;
  __bf16* hbuf  = (__bf16*)(ws + 0);          //  8 MB  LN1/LN2 output (bf16)
  __bf16* wqkvT = (__bf16*)(ws + 8 * MB);     //  6 MB
  __bf16* woutT = (__bf16*)(ws + 14 * MB);    //  2 MB
  __bf16* wfc1T = (__bf16*)(ws + 16 * MB);    //  8 MB
  __bf16* wfc2T = (__bf16*)(ws + 24 * MB);    //  8 MB
  __bf16* qbuf  = (__bf16*)(ws + 32 * MB);    //  8 MB  (B,H,T,D)
  __bf16* kbuf  = (__bf16*)(ws + 40 * MB);    //  8 MB  (B,H,T,D)
  __bf16* vtbuf = (__bf16*)(ws + 48 * MB);    //  8 MB  (B,H,D,T)
  __bf16* attnO = (__bf16*)(ws + 56 * MB);    //  8 MB  (B,T,C)
  __bf16* hid   = (__bf16*)(ws + 32 * MB);    // 32 MB  reuses q/k/vT/attnO (dead by FC1)
  float*  x1    = (float*)(ws + 64 * MB);     // 16 MB  residual stream after attention
  __bf16* memK  = (__bf16*)(ws + 80 * MB);            // 128 KB (H,M,D)
  __bf16* memVT = (__bf16*)(ws + 80 * MB + 256 * 1024); // 128 KB (H,D,M)

  // 1. LN1 -> bf16 activations
  ln_kernel<<<4096, 256, 0, stream>>>(x, ln1_g, ln1_b, hbuf);

  // 2. weight transpose+convert (N-major bf16 for WMMA B operand)
  transpose_bf16<<<dim3(3072 / 16, 1024 / 16), 256, 0, stream>>>(w_qkv, wqkvT, 1024, 3072);
  transpose_bf16<<<dim3(1024 / 16, 1024 / 16), 256, 0, stream>>>(w_out, woutT, 1024, 1024);
  transpose_bf16<<<dim3(4096 / 16, 1024 / 16), 256, 0, stream>>>(w_fc1, wfc1T, 1024, 4096);
  transpose_bf16<<<dim3(1024 / 16, 4096 / 16), 256, 0, stream>>>(w_fc2, wfc2T, 4096, 1024);

  // 3. compressive memory projection (tiny: 64x1024)
  memproj_kernel<<<64, 256, 0, stream>>>(memory, w_mem, b_mem, memK, memVT);

  // 4. QKV GEMM (4096 x 3072 x 1024) with Q/K/V^T-split epilogue
  { EpiQKV e{b_qkv, qbuf, kbuf, vtbuf};
    gemm_bf16<1024, EpiQKV><<<dim3(3072 / 128, 4096 / 64), 256, 0, stream>>>(hbuf, wqkvT, e); }

  // 5. flash attention (local) + memory attention, summed
  attn_kernel<<<dim3(2048 / 128, 32), 256, 0, stream>>>(qbuf, kbuf, vtbuf, memK, memVT, attnO);

  // 6. out projection + bias + residual -> x1 (fp32)
  { EpiBiasResAdd e{b_out, x, x1};
    gemm_bf16<1024, EpiBiasResAdd><<<dim3(1024 / 128, 4096 / 64), 256, 0, stream>>>(attnO, woutT, e); }

  // 7. LN2 -> bf16 (reuses hbuf)
  ln_kernel<<<4096, 256, 0, stream>>>(x1, ln2_g, ln2_b, hbuf);

  // 8. FC1 (4096 x 4096 x 1024) + exact GELU -> bf16 hidden
  { EpiGelu e{b_fc1, hid};
    gemm_bf16<1024, EpiGelu><<<dim3(4096 / 128, 4096 / 64), 256, 0, stream>>>(hbuf, wfc1T, e); }

  // 9. FC2 (4096 x 1024 x 4096) + bias + residual -> final output (fp32)
  { EpiBiasResAdd e{b_fc2, x1, out};
    gemm_bf16<4096, EpiBiasResAdd><<<dim3(1024 / 128, 4096 / 64), 256, 0, stream>>>(hid, wfc2T, e); }
}
